// SSGCN_22531398434878
// MI455X (gfx1250) — compile-verified, matrix operand order checked by reading, not run on hardware
//
#include <hip/hip_runtime.h>
#include <hip/hip_bf16.h>
#include <cstdint>
#include <cstddef>

// ---------------------------------------------------------------------------
// Problem dims (compile-time)
// ---------------------------------------------------------------------------
#define BB   64      // batch
#define SS   256     // seq len
#define HH   768     // hidden (2*Hh)
#define HHH  384     // LSTM hidden per dir
#define GG   1536    // 4*Hh gate width
#define EE   4       // edges per batch
#define AA   4       // aspect positions
#define CC   3       // classes

typedef __attribute__((ext_vector_type(16))) __bf16 v16bf;
typedef __attribute__((ext_vector_type(8)))  __bf16 v8bf;
typedef __attribute__((ext_vector_type(8)))  float  v8f;

__device__ __forceinline__ float sigf(float x) { return 1.0f / (1.0f + __expf(-x)); }

// Build a WMMA 16-bit A/B fragment from a row pointer (row-major, contiguous K).
// Per ISA layout: lanes 0-15 hold K chunks {0..7, 16..23}, lanes 16-31 hold
// {8..15, 24..31}; half8 = (lane>>4)*8.
__device__ __forceinline__ v16bf frag16(const __bf16* __restrict__ p, int half8) {
    v8bf lo = *(const v8bf*)(p + half8);
    v8bf hi = *(const v8bf*)(p + 16 + half8);
    return __builtin_shufflevector(lo, hi, 0, 1, 2, 3, 4, 5, 6, 7,
                                            8, 9, 10, 11, 12, 13, 14, 15);
}

// ---------------------------------------------------------------------------
// fp32 -> bf16 flat cast / zero fill
// ---------------------------------------------------------------------------
__global__ void k_cast_bf16(const float* __restrict__ src, __bf16* __restrict__ dst, size_t n) {
    for (size_t i = (size_t)blockIdx.x * blockDim.x + threadIdx.x; i < n;
         i += (size_t)gridDim.x * blockDim.x)
        dst[i] = (__bf16)src[i];
}

__global__ void k_zero_bf16(__bf16* __restrict__ dst, size_t n) {
    for (size_t i = (size_t)blockIdx.x * blockDim.x + threadIdx.x; i < n;
         i += (size_t)gridDim.x * blockDim.x)
        dst[i] = (__bf16)0.0f;
}

// bias sum: out = a + b  (fp32)
__global__ void k_bias2(const float* __restrict__ a, const float* __restrict__ b,
                        float* __restrict__ out, int n) {
    int i = blockIdx.x * blockDim.x + threadIdx.x;
    if (i < n) out[i] = a[i] + b[i];
}

// Permute+cast bert_out (B,S,H) fp32 -> Xf[t*64+b][h], Xb[t*64+b][h] (bf16)
__global__ void k_permx(const float* __restrict__ x, __bf16* __restrict__ xf,
                        __bf16* __restrict__ xb) {
    size_t n = (size_t)BB * SS * HH;
    for (size_t i = (size_t)blockIdx.x * blockDim.x + threadIdx.x; i < n;
         i += (size_t)gridDim.x * blockDim.x) {
        int h = (int)(i % HH);
        size_t r = i / HH;
        int b = (int)(r % BB);
        int t = (int)(r / BB);
        xf[i] = (__bf16)x[((size_t)b * SS + t) * HH + h];
        xb[i] = (__bf16)x[((size_t)b * SS + (SS - 1 - t)) * HH + h];
    }
}

// ---------------------------------------------------------------------------
// WMMA bf16 GEMM:  C[M][N] = A(MxK bf16 row-major) * B(NxK bf16)^T + bias[N]
// Requirements: M % 64 == 0, K % 64 == 0, B padded to ceil(N/64)*64 rows.
// Block: 256 threads (8 waves). 64x64 tile, staged K-chunks of 64.
// ---------------------------------------------------------------------------
__global__ __launch_bounds__(256)
void k_gemm_bf16(const __bf16* __restrict__ A, const __bf16* __restrict__ Bm,
                 const float* __restrict__ bias, float* __restrict__ Cf,
                 __bf16* __restrict__ Cbf, int M, int N, int K, int relu) {
    __shared__ alignas(16) __bf16 As[64][64];
    __shared__ alignas(16) __bf16 Bs[64][64];

    const int m0   = blockIdx.y * 64;
    const int n0   = blockIdx.x * 64;
    const int tid  = threadIdx.x;
    const int lane = tid & 31;
    const int wv   = tid >> 5;
    const int mt   = wv & 3;            // m sub-tile 0..3
    const int nt0  = (wv >> 2) * 2;     // n sub-tiles {0,1} or {2,3}

    v8f acc0 = {0.f, 0.f, 0.f, 0.f, 0.f, 0.f, 0.f, 0.f};
    v8f acc1 = {0.f, 0.f, 0.f, 0.f, 0.f, 0.f, 0.f, 0.f};

    // staging: thread -> (row = tid>>2, cols (tid&3)*16 .. +15), two 16B chunks
    const int sr = tid >> 2;
    const int sc = (tid & 3) * 16;
    const __bf16* aptr = A + (size_t)(m0 + sr) * K + sc;
    const __bf16* bptr = Bm + (size_t)(n0 + sr) * K + sc;

    const int half8 = (lane >> 4) * 8;
    const int mr  = mt * 16 + (lane & 15);
    const int nr0 = nt0 * 16 + (lane & 15);
    const int nr1 = nr0 + 16;

    for (int k0 = 0; k0 < K; k0 += 64) {
        *(uint4*)&As[sr][sc]     = *(const uint4*)(aptr);
        *(uint4*)&As[sr][sc + 8] = *(const uint4*)(aptr + 8);
        *(uint4*)&Bs[sr][sc]     = *(const uint4*)(bptr);
        *(uint4*)&Bs[sr][sc + 8] = *(const uint4*)(bptr + 8);
        aptr += 64;
        bptr += 64;
        __syncthreads();
#pragma unroll
        for (int kk = 0; kk < 64; kk += 32) {
            v16bf a  = frag16(&As[mr][kk], half8);
            v16bf b0 = frag16(&Bs[nr0][kk], half8);
            acc0 = __builtin_amdgcn_wmma_f32_16x16x32_bf16(
                false, a, false, b0, (short)0, acc0, false, false);
            v16bf b1 = frag16(&Bs[nr1][kk], half8);
            acc1 = __builtin_amdgcn_wmma_f32_16x16x32_bf16(
                false, a, false, b1, (short)0, acc1, false, false);
        }
        __syncthreads();
    }

    // epilogue: element (e, lane) of acc -> row m0+mt*16+e+8*(lane>>4)
    const int halfo = (lane >> 4) * 8;
    {
        int col = n0 + nt0 * 16 + (lane & 15);
        if (col < N) {
            float bv = bias ? bias[col] : 0.0f;
#pragma unroll
            for (int e = 0; e < 8; ++e) {
                int row = m0 + mt * 16 + e + halfo;
                float v = acc0[e] + bv;
                if (relu) v = fmaxf(v, 0.0f);
                size_t o = (size_t)row * N + col;
                if (Cf)  Cf[o]  = v;
                if (Cbf) Cbf[o] = (__bf16)v;
            }
        }
        col += 16;
        if (col < N) {
            float bv = bias ? bias[col] : 0.0f;
#pragma unroll
            for (int e = 0; e < 8; ++e) {
                int row = m0 + mt * 16 + e + halfo;
                float v = acc1[e] + bv;
                if (relu) v = fmaxf(v, 0.0f);
                size_t o = (size_t)row * N + col;
                if (Cf)  Cf[o]  = v;
                if (Cbf) Cbf[o] = (__bf16)v;
            }
        }
    }
}

// ---------------------------------------------------------------------------
// Persistent BiLSTM recurrence. grid.x = 2 (dir), block = 1024 (32 waves).
// xg: [S][64][1536] bf16 (input proj + biases, per dir), whh: [1536][384] bf16.
// h state in LDS (bf16), c state in VGPRs. Per step:
//   G(64x1536) = Hprev(64x384) * whh^T   -> 96 16x16 tile-tasks, 3 per wave.
// ---------------------------------------------------------------------------
__global__ __launch_bounds__(1024)
void k_lstm(const __bf16* __restrict__ xg_f, const __bf16* __restrict__ xg_b,
            const __bf16* __restrict__ whh_f, const __bf16* __restrict__ whh_b,
            float* __restrict__ hout) {
    const int dir = blockIdx.x;
    const __bf16* __restrict__ xg  = dir ? xg_b : xg_f;
    const __bf16* __restrict__ whh = dir ? whh_b : whh_f;

    __shared__ alignas(16) __bf16 hbuf[BB][HHH];   // 48 KB

    const int tid   = threadIdx.x;
    const int lane  = tid & 31;
    const int wv    = tid >> 5;
    const int half8 = (lane >> 4) * 8;
    const int halfo = (lane >> 4) * 8;

    for (int i = tid; i < BB * HHH; i += 1024) ((__bf16*)hbuf)[i] = (__bf16)0.0f;

    v8f cst[3];
#pragma unroll
    for (int j = 0; j < 3; ++j)
        cst[j] = (v8f){0.f, 0.f, 0.f, 0.f, 0.f, 0.f, 0.f, 0.f};

    __syncthreads();

    for (int t = 0; t < SS; ++t) {
        float hn[3][8];
#pragma unroll
        for (int j = 0; j < 3; ++j) {
            const int tk = wv * 3 + j;      // 0..95
            const int mt = tk / 24;         // 0..3  (batch tile)
            const int nh = tk % 24;         // 0..23 (hidden tile)
            const int mr = mt * 16 + (lane & 15);
            const int nr = nh * 16 + (lane & 15);

            v8f g0 = {0.f, 0.f, 0.f, 0.f, 0.f, 0.f, 0.f, 0.f};
            v8f g1 = g0, g2 = g0, g3 = g0;

            const __bf16* w0 = whh + (size_t)(0 * HHH + nr) * HHH;
            const __bf16* w1 = whh + (size_t)(1 * HHH + nr) * HHH;
            const __bf16* w2 = whh + (size_t)(2 * HHH + nr) * HHH;
            const __bf16* w3 = whh + (size_t)(3 * HHH + nr) * HHH;

            for (int k = 0; k < HHH; k += 32) {
                v16bf a = frag16(&hbuf[mr][k], half8);
                __builtin_prefetch(w0 + k + 32, 0, 1);
                g0 = __builtin_amdgcn_wmma_f32_16x16x32_bf16(
                    false, a, false, frag16(w0 + k, half8), (short)0, g0, false, false);
                g1 = __builtin_amdgcn_wmma_f32_16x16x32_bf16(
                    false, a, false, frag16(w1 + k, half8), (short)0, g1, false, false);
                g2 = __builtin_amdgcn_wmma_f32_16x16x32_bf16(
                    false, a, false, frag16(w2 + k, half8), (short)0, g2, false, false);
                g3 = __builtin_amdgcn_wmma_f32_16x16x32_bf16(
                    false, a, false, frag16(w3 + k, half8), (short)0, g3, false, false);
            }

            const int col  = nh * 16 + (lane & 15);
            const int srow = dir ? (SS - 1 - t) : t;
#pragma unroll
            for (int e = 0; e < 8; ++e) {
                const int b = mt * 16 + e + halfo;
                const __bf16* xr = xg + ((size_t)t * BB + b) * GG;
                float iv = g0[e] + (float)xr[col];
                float fv = g1[e] + (float)xr[HHH + col];
                float gv = g2[e] + (float)xr[2 * HHH + col];
                float ov = g3[e] + (float)xr[3 * HHH + col];
                float cn = sigf(fv) * cst[j][e] + sigf(iv) * tanhf(gv);
                float hv = sigf(ov) * tanhf(cn);
                cst[j][e] = cn;
                hn[j][e] = hv;
                hout[((size_t)b * SS + srow) * HH + dir * HHH + col] = hv;
            }
        }
        __syncthreads();   // all reads of hbuf done
#pragma unroll
        for (int j = 0; j < 3; ++j) {
            const int tk = wv * 3 + j;
            const int mt = tk / 24, nh = tk % 24;
            const int col = nh * 16 + (lane & 15);
#pragma unroll
            for (int e = 0; e < 8; ++e) {
                const int b = mt * 16 + e + halfo;
                hbuf[b][col] = (__bf16)hn[j][e];
            }
        }
        __syncthreads();   // new h visible
    }
}

// ---------------------------------------------------------------------------
// Mean over aspect positions: out[b][:] = mean_a x[b][pos[b][a]][:]
// ---------------------------------------------------------------------------
__global__ __launch_bounds__(256)
void k_aspect_mean(const float* __restrict__ x, const int* __restrict__ pos,
                   float* __restrict__ out, __bf16* __restrict__ out_bf) {
    const int b = blockIdx.x;
    for (int c = threadIdx.x; c < HH; c += 256) {
        float s = 0.0f;
#pragma unroll
        for (int a = 0; a < AA; ++a) {
            int p = pos[b * AA + a];
            s += x[((size_t)b * SS + p) * HH + c];
        }
        s *= (1.0f / AA);
        out[(size_t)b * HH + c] = s;
        if (out_bf) out_bf[(size_t)b * HH + c] = (__bf16)s;
    }
}

// ---------------------------------------------------------------------------
// Span attention: scores = h.q, softmax over s, span = sum attn*h
// ---------------------------------------------------------------------------
__global__ __launch_bounds__(256)
void k_attn(const float* __restrict__ h, const float* __restrict__ q,
            float* __restrict__ span) {
    const int b = blockIdx.x, tid = threadIdx.x;
    __shared__ float qs[HH];
    __shared__ float sc[SS];
    __shared__ float red[256];

    for (int i = tid; i < HH; i += 256) qs[i] = q[(size_t)b * HH + i];
    __syncthreads();

    const float* hr = h + ((size_t)b * SS + tid) * HH;
    float d = 0.0f;
    for (int i = 0; i < HH; ++i) d += hr[i] * qs[i];
    sc[tid] = d;
    red[tid] = d;
    __syncthreads();
    for (int off = 128; off > 0; off >>= 1) {
        if (tid < off) red[tid] = fmaxf(red[tid], red[tid + off]);
        __syncthreads();
    }
    const float mx = red[0];
    __syncthreads();
    float e = __expf(sc[tid] - mx);
    sc[tid] = e;
    red[tid] = e;
    __syncthreads();
    for (int off = 128; off > 0; off >>= 1) {
        if (tid < off) red[tid] += red[tid + off];
        __syncthreads();
    }
    const float inv = 1.0f / red[0];
    __syncthreads();
    sc[tid] *= inv;
    __syncthreads();

    for (int c = tid; c < HH; c += 256) {
        float s = 0.0f;
        for (int t = 0; t < SS; ++t)
            s += sc[t] * h[((size_t)b * SS + t) * HH + c];
        span[(size_t)b * HH + c] = s;
    }
}

// ---------------------------------------------------------------------------
// GCN normalized aggregation (per batch). one block per b, thread = node.
// ---------------------------------------------------------------------------
__global__ __launch_bounds__(256)
void k_gcn_agg(const float* __restrict__ xw, const int* __restrict__ eidx,
               const float* __restrict__ ew, const float* __restrict__ bias,
               float* __restrict__ outf, __bf16* __restrict__ outbf, int relu) {
    const int b = blockIdx.x, d = threadIdx.x;
    __shared__ float deg[SS];
    __shared__ int   esrc[EE], edst[EE];
    __shared__ float ewt[EE];
    __shared__ float dinv[SS];

    deg[d] = 1.0f;  // self-loop weight
    if (d < EE) {
        esrc[d] = eidx[(size_t)b * 2 * EE + d];
        edst[d] = eidx[(size_t)b * 2 * EE + EE + d];
        ewt[d]  = ew[(size_t)b * EE + d];
    }
    __syncthreads();
    if (d < EE) atomicAdd(&deg[edst[d]], ewt[d]);
    __syncthreads();
    dinv[d] = rsqrtf(deg[d]);
    __syncthreads();

    float coef[EE];
    int   nsrc[EE];
    int   nE = 0;
#pragma unroll
    for (int e = 0; e < EE; ++e)
        if (edst[e] == d) { coef[nE] = dinv[esrc[e]] * ewt[e] * dinv[d]; nsrc[nE] = esrc[e]; ++nE; }
    const float wself = dinv[d] * dinv[d];

    const float* xr = xw + ((size_t)b * SS + d) * HH;
    for (int c = 0; c < HH; ++c) {
        float s = wself * xr[c];
        for (int e = 0; e < nE; ++e)
            s += coef[e] * xw[((size_t)b * SS + nsrc[e]) * HH + c];
        s += bias[c];
        if (relu) s = fmaxf(s, 0.0f);
        size_t o = ((size_t)b * SS + d) * HH + c;
        if (outf)  outf[o]  = s;
        if (outbf) outbf[o] = (__bf16)s;
    }
}

// concat(span, syn) -> bf16 (64 x 1536)
__global__ void k_cat_bf(const float* __restrict__ span, const float* __restrict__ syn,
                         __bf16* __restrict__ cat) {
    int i = blockIdx.x * 256 + threadIdx.x;
    if (i < BB * 2 * HH) {
        int b = i / (2 * HH), c = i % (2 * HH);
        float v = (c < HH) ? span[(size_t)b * HH + c] : syn[(size_t)b * HH + (c - HH)];
        cat[i] = (__bf16)v;
    }
}

// gate = sigmoid(glin); fused = gate*syn + (1-gate)*span  -> bf16
__global__ void k_fuse(const float* __restrict__ glin, const float* __restrict__ span,
                       const float* __restrict__ syn, __bf16* __restrict__ fused_bf) {
    int i = blockIdx.x * 256 + threadIdx.x;
    if (i < BB * HH) {
        float g = sigf(glin[i]);
        fused_bf[i] = (__bf16)(g * syn[i] + (1.0f - g) * span[i]);
    }
}

// ---------------------------------------------------------------------------
// Host orchestration
// ---------------------------------------------------------------------------
extern "C" void kernel_launch(void* const* d_in, const int* in_sizes, int n_in,
                              void* d_out, int out_size, void* d_ws, size_t ws_size,
                              hipStream_t stream) {
    const float* bert_out = (const float*)d_in[0];
    const float* edge_w   = (const float*)d_in[1];
    const float* w_ih_f   = (const float*)d_in[2];
    const float* w_hh_f   = (const float*)d_in[3];
    const float* b_ih_f   = (const float*)d_in[4];
    const float* b_hh_f   = (const float*)d_in[5];
    const float* w_ih_b   = (const float*)d_in[6];
    const float* w_hh_b   = (const float*)d_in[7];
    const float* b_ih_b   = (const float*)d_in[8];
    const float* b_hh_b   = (const float*)d_in[9];
    const float* span_w   = (const float*)d_in[10];
    const float* span_b   = (const float*)d_in[11];
    const float* gcn1_w   = (const float*)d_in[12];
    const float* gcn1_b   = (const float*)d_in[13];
    const float* gcn2_w   = (const float*)d_in[14];
    const float* gcn2_b   = (const float*)d_in[15];
    const float* gate_w   = (const float*)d_in[16];
    const float* gate_b   = (const float*)d_in[17];
    const float* cls_w    = (const float*)d_in[18];
    const float* cls_b    = (const float*)d_in[19];
    const int*   edge_idx = (const int*)d_in[20];
    const int*   asp_pos  = (const int*)d_in[21];
    float* out = (float*)d_out;

    size_t off = 0;
    auto alloc = [&](size_t bytes) -> void* {
        void* p = (char*)d_ws + off;
        off += (bytes + 255) & ~(size_t)255;
        return p;
    };
    const size_t MS = (size_t)BB * SS;             // 16384 rows

    __bf16* Xf       = (__bf16*)alloc(MS * HH * 2);
    __bf16* Xb       = (__bf16*)alloc(MS * HH * 2);
    __bf16* wihf_bf  = (__bf16*)alloc((size_t)GG * HH * 2);
    __bf16* wihb_bf  = (__bf16*)alloc((size_t)GG * HH * 2);
    __bf16* whhf_bf  = (__bf16*)alloc((size_t)GG * HHH * 2);
    __bf16* whhb_bf  = (__bf16*)alloc((size_t)GG * HHH * 2);
    float*  biasf    = (float*)alloc(GG * 4);
    float*  biasb    = (float*)alloc(GG * 4);
    __bf16* xgf      = (__bf16*)alloc((size_t)SS * BB * GG * 2);
    __bf16* xgb      = (__bf16*)alloc((size_t)SS * BB * GG * 2);
    float*  h        = (float*)alloc(MS * HH * 4);
    __bf16* h_bf     = (__bf16*)alloc(MS * HH * 2);
    __bf16* spanw_bf = (__bf16*)alloc((size_t)HH * HH * 2);
    __bf16* gcn1w_bf = (__bf16*)alloc((size_t)HH * HH * 2);
    __bf16* gcn2w_bf = (__bf16*)alloc((size_t)HH * HH * 2);
    __bf16* gatew_bf = (__bf16*)alloc((size_t)HH * 2 * HH * 2);
    __bf16* clsw_bf  = (__bf16*)alloc((size_t)64 * HH * 2);   // padded to 64 rows
    float*  avec     = (float*)alloc((size_t)BB * HH * 4);
    __bf16* avec_bf  = (__bf16*)alloc((size_t)BB * HH * 2);
    float*  q        = (float*)alloc((size_t)BB * HH * 4);
    float*  span     = (float*)alloc((size_t)BB * HH * 4);
    float*  syn      = (float*)alloc((size_t)BB * HH * 4);
    float*  xw       = (float*)alloc(MS * HH * 4);
    __bf16* x1_bf    = (__bf16*)alloc(MS * HH * 2);
    float*  x2       = (float*)alloc(MS * HH * 4);
    __bf16* cat_bf   = (__bf16*)alloc((size_t)BB * 2 * HH * 2);
    float*  glin     = (float*)alloc((size_t)BB * HH * 4);
    __bf16* fused_bf = (__bf16*)alloc((size_t)BB * HH * 2);
    (void)ws_size; (void)n_in; (void)in_sizes; (void)out_size;

    auto cast = [&](const float* s, __bf16* d, size_t n) {
        int blocks = (int)((n + 255) / 256);
        if (blocks > 65535) blocks = 65535;
        k_cast_bf16<<<blocks, 256, 0, stream>>>(s, d, n);
    };

    // 1) weight casts (cls_w padded: zero 64x768 then cast 3x768 into it)
    cast(w_ih_f, wihf_bf, (size_t)GG * HH);
    cast(w_ih_b, wihb_bf, (size_t)GG * HH);
    cast(w_hh_f, whhf_bf, (size_t)GG * HHH);
    cast(w_hh_b, whhb_bf, (size_t)GG * HHH);
    cast(span_w, spanw_bf, (size_t)HH * HH);
    cast(gcn1_w, gcn1w_bf, (size_t)HH * HH);
    cast(gcn2_w, gcn2w_bf, (size_t)HH * HH);
    cast(gate_w, gatew_bf, (size_t)HH * 2 * HH);
    k_zero_bf16<<<(64 * HH + 255) / 256, 256, 0, stream>>>(clsw_bf, (size_t)64 * HH);
    cast(cls_w, clsw_bf, (size_t)CC * HH);
    k_bias2<<<(GG + 255) / 256, 256, 0, stream>>>(b_ih_f, b_hh_f, biasf, GG);
    k_bias2<<<(GG + 255) / 256, 256, 0, stream>>>(b_ih_b, b_hh_b, biasb, GG);

    // 2) permute+cast x
    k_permx<<<8192, 256, 0, stream>>>(bert_out, Xf, Xb);

    // 3) LSTM input projections: xg = X * w_ih^T + (b_ih+b_hh)   [16384 x 1536]
    {
        dim3 g(GG / 64, (unsigned)(MS / 64));
        k_gemm_bf16<<<g, 256, 0, stream>>>(Xf, wihf_bf, biasf, nullptr, xgf,
                                           (int)MS, GG, HH, 0);
        k_gemm_bf16<<<g, 256, 0, stream>>>(Xb, wihb_bf, biasb, nullptr, xgb,
                                           (int)MS, GG, HH, 0);
    }

    // 4) BiLSTM recurrence (persistent WMMA, one workgroup per direction)
    k_lstm<<<2, 1024, 0, stream>>>(xgf, xgb, whhf_bf, whhb_bf, h);

    // 5) h -> bf16 for GCN GEMMs
    cast(h, h_bf, MS * HH);

    // 6) aspect vector + q = avec @ span_w^T + span_b
    k_aspect_mean<<<BB, 256, 0, stream>>>(h, asp_pos, avec, avec_bf);
    {
        dim3 g(HH / 64, BB / 64);
        k_gemm_bf16<<<g, 256, 0, stream>>>(avec_bf, spanw_bf, span_b, q, nullptr,
                                           BB, HH, HH, 0);
    }

    // 7) span attention
    k_attn<<<BB, 256, 0, stream>>>(h, q, span);

    // 8) GCN layer 1: xw = h @ W1^T ; aggregate + relu -> x1 (bf16)
    {
        dim3 g(HH / 64, (unsigned)(MS / 64));
        k_gemm_bf16<<<g, 256, 0, stream>>>(h_bf, gcn1w_bf, nullptr, xw, nullptr,
                                           (int)MS, HH, HH, 0);
    }
    k_gcn_agg<<<BB, 256, 0, stream>>>(xw, edge_idx, edge_w, gcn1_b,
                                      nullptr, x1_bf, 1);

    // 9) GCN layer 2
    {
        dim3 g(HH / 64, (unsigned)(MS / 64));
        k_gemm_bf16<<<g, 256, 0, stream>>>(x1_bf, gcn2w_bf, nullptr, xw, nullptr,
                                           (int)MS, HH, HH, 0);
    }
    k_gcn_agg<<<BB, 256, 0, stream>>>(xw, edge_idx, edge_w, gcn2_b,
                                      x2, nullptr, 0);

    // 10) syn, gate, fuse, classify
    k_aspect_mean<<<BB, 256, 0, stream>>>(x2, asp_pos, syn, nullptr);
    k_cat_bf<<<(BB * 2 * HH + 255) / 256, 256, 0, stream>>>(span, syn, cat_bf);
    {
        dim3 g(HH / 64, BB / 64);
        k_gemm_bf16<<<g, 256, 0, stream>>>(cat_bf, gatew_bf, gate_b, glin, nullptr,
                                           BB, HH, 2 * HH, 0);
    }
    k_fuse<<<(BB * HH + 255) / 256, 256, 0, stream>>>(glin, span, syn, fused_bf);
    {
        dim3 g(1, BB / 64);
        k_gemm_bf16<<<g, 256, 0, stream>>>(fused_bf, clsw_bf, cls_b, out, nullptr,
                                           BB, CC, HH, 0);
    }
}
